// HRMLM_72438918414656
// MI455X (gfx1250) — matrix operand order repaired and checked by reading, not compiled, not run on hardware
//
#include <hip/hip_runtime.h>
#include <hip/hip_bf16.h>

// ---- model dims ----
#define V_SZ   32000
#define E_SZ   768
#define HL_SZ  1024
#define HH_SZ  768
#define B_SZ   8
#define S_SZ   256
#define EPS_LN 1e-5f

// ---- CDNA5 WMMA types ----
typedef __attribute__((ext_vector_type(16))) __bf16       v16bf;
typedef __attribute__((ext_vector_type(8)))  float        v8f;
typedef __attribute__((ext_vector_type(4)))  unsigned int u32x4;

union Frag16 { u32x4 u[2]; v16bf v; };

// ---------------- helpers ----------------
__device__ __forceinline__ float block_sum(float v, float* red) {
    int tid = threadIdx.x;
    red[tid] = v;
    __syncthreads();
    for (int s = 128; s > 0; s >>= 1) {
        if (tid < s) red[tid] += red[tid + s];
        __syncthreads();
    }
    float r = red[0];
    __syncthreads();
    return r;
}

// Core WMMA k-loop. arow must already include +hi*8, brow must include +hi*8.
// Unpredicated, fully uniform: 4x b128 load + prefetch + wmma per 32-deep k-step.
__device__ __forceinline__ v8f gemm_k(const __bf16* __restrict__ arow,
                                      const __bf16* __restrict__ brow,
                                      int K, v8f acc) {
    for (int kb = 0; kb < K; kb += 32) {
        Frag16 a, b;
        a.u[0] = *reinterpret_cast<const u32x4*>(arow + kb);       // K = kb+hi*8 .. +7
        a.u[1] = *reinterpret_cast<const u32x4*>(arow + kb + 16);  // K = kb+hi*8+16 ..
        b.u[0] = *reinterpret_cast<const u32x4*>(brow + kb);       // K = kb+hi*8 ..
        b.u[1] = *reinterpret_cast<const u32x4*>(brow + kb + 16);
        __builtin_prefetch((const void*)(brow + kb + 32), 0, 3);   // near-scope stream
        acc = __builtin_amdgcn_wmma_f32_16x16x32_bf16(false, a.v, false, b.v,
                                                      (short)0, acc, false, false);
    }
    return acc;
}

__device__ __forceinline__ void store_tile(float* __restrict__ outp, int Ntot,
                                           int n0, int lane, v8f acc,
                                           const float* __restrict__ bias) {
    int col = n0 + (lane & 15);
    float bv = bias[col];
    int rbase = (lane >> 4) * 8;
#pragma unroll
    for (int r = 0; r < 8; ++r)
        outp[(long)(rbase + r) * Ntot + col] = acc[r] + bv;
}

// ---------------- zero fill ----------------
__global__ __launch_bounds__(256) void k_zero(unsigned int* p, long n) {
    long i = (long)blockIdx.x * 256 + threadIdx.x;
    if (i < n) p[i] = 0u;
}

// ---------------- f32 [slices][K][N] -> bf16 transposed [slices][N][K] ----------------
__global__ __launch_bounds__(256) void k_w2bf16t(const float* __restrict__ src,
                                                 __bf16* __restrict__ dst,
                                                 int K, int N, int slices) {
    long per = (long)K * N;
    long total = per * slices;
    for (long i = (long)blockIdx.x * 256 + threadIdx.x; i < total;
         i += (long)gridDim.x * 256) {
        long sl = i / per;
        long r  = i - sl * per;
        long n  = r / K;
        long k  = r - n * K;
        dst[i] = (__bf16)src[sl * per + k * N + n];
    }
}

// ---------------- x = LN(emb[ids]) -> bf16 X[s][16][768] (rows 8..15 stay zero) -------
__global__ __launch_bounds__(256) void k_embed(const int* __restrict__ ids,
                                               const float* __restrict__ emb,
                                               const float* __restrict__ g,
                                               const float* __restrict__ bt,
                                               __bf16* __restrict__ X) {
    __shared__ float red[256];
    int s = blockIdx.x >> 3, b = blockIdx.x & 7;
    int tid = threadIdx.x;
    int id = ids[b * S_SZ + s];
    const float* row = emb + (long)id * E_SZ;
    float sm = 0.f, sq = 0.f;
    for (int i = tid; i < E_SZ; i += 256) { float v = row[i]; sm += v; sq += v * v; }
    sm = block_sum(sm, red);
    sq = block_sum(sq, red);
    float mean = sm / E_SZ;
    float var  = sq / E_SZ - mean * mean;
    float rstd = rsqrtf(var + EPS_LN);
    __bf16* dst = X + ((long)s * 16 + b) * E_SZ;
    for (int i = tid; i < E_SZ; i += 256)
        dst[i] = (__bf16)((row[i] - mean) * rstd * g[i] + bt[i]);
}

// ---------------- fused GRU-cell GEMM pair -------------------------------------------
// Tiles [0, Ntot/16)           : xi = concat(Ax1[16xKx1], Ax2[16xKx2]) @ WTi^T + bi
// Tiles [Ntot/16, 2*Ntot/16)   : hi = Ah[16xKh] @ WTh^T + bh
// All A buffers are 16 rows (rows 8..15 pre-zeroed) -> no predication anywhere.
__global__ __launch_bounds__(256) void k_gru_gemms(
        const __bf16* __restrict__ Ax1, int Kx1,
        const __bf16* __restrict__ Ax2, int Kx2,
        const __bf16* __restrict__ WTi, const float* __restrict__ bi,
        float* __restrict__ outXi,
        const __bf16* __restrict__ Ah, int Kh,
        const __bf16* __restrict__ WTh, const float* __restrict__ bh,
        float* __restrict__ outHi,
        int Ntot) {
    int wave = threadIdx.x >> 5, lane = threadIdx.x & 31;
    int ntiles = Ntot >> 4;
    int tile = blockIdx.x * 8 + wave;
    int m = lane & 15, hi = lane >> 4;
    v8f acc = {};
    if (tile < ntiles) {                       // ---- xi GEMM (with concat) ----
        int n0 = tile << 4;
        int Kx = Kx1 + Kx2;
        const __bf16* brow = WTi + (long)(n0 + m) * Kx + hi * 8;
        acc = gemm_k(Ax1 + (long)m * Kx1 + hi * 8, brow, Kx1, acc);
        if (Kx2 > 0)
            acc = gemm_k(Ax2 + (long)m * Kx2 + hi * 8, brow + Kx1, Kx2, acc);
        store_tile(outXi, Ntot, n0, lane, acc, bi);
    } else if (tile < 2 * ntiles) {            // ---- hi GEMM ----
        int n0 = (tile - ntiles) << 4;
        const __bf16* brow = WTh + (long)(n0 + m) * Kh + hi * 8;
        acc = gemm_k(Ah + (long)m * Kh + hi * 8, brow, Kh, acc);
        store_tile(outHi, Ntot, n0, lane, acc, bh);
    }
}

// ---------------- LN(gates) + GRU state update ----------------
// xi_raw/hi_raw: [16][3H] f32 (rows 8..15 junk, never read). One block per batch row.
__global__ __launch_bounds__(256) void k_gru_update(const float* __restrict__ xi_raw,
                                                    const float* __restrict__ hi_raw,
                                                    const float* __restrict__ lni_g,
                                                    const float* __restrict__ lni_b,
                                                    const float* __restrict__ lnh_g,
                                                    const float* __restrict__ lnh_b,
                                                    float* __restrict__ h,
                                                    __bf16* __restrict__ hA,
                                                    __bf16* __restrict__ hA2,
                                                    int H) {
    __shared__ float red[256];
    __shared__ float mu[6], rs[6];
    int b = blockIdx.x, tid = threadIdx.x;
    int s3 = 3 * H;
    for (int j = 0; j < 6; ++j) {
        const float* rrow = (j < 3) ? (xi_raw + (long)b * s3 + j * H)
                                    : (hi_raw + (long)b * s3 + (j - 3) * H);
        float sm = 0.f, sq = 0.f;
        for (int i = tid; i < H; i += 256) { float v = rrow[i]; sm += v; sq += v * v; }
        sm = block_sum(sm, red);
        sq = block_sum(sq, red);
        if (tid == 0) {
            float m = sm / H;
            mu[j] = m;
            rs[j] = rsqrtf(sq / H - m * m + EPS_LN);
        }
    }
    __syncthreads();
    const float* xrow = xi_raw + (long)b * s3;
    const float* hrow = hi_raw + (long)b * s3;
    for (int i = tid; i < H; i += 256) {
        float xr = (xrow[i]         - mu[0]) * rs[0] * lni_g[i]         + lni_b[i];
        float xz = (xrow[H + i]     - mu[1]) * rs[1] * lni_g[H + i]     + lni_b[H + i];
        float xn = (xrow[2 * H + i] - mu[2]) * rs[2] * lni_g[2 * H + i] + lni_b[2 * H + i];
        float hr = (hrow[i]         - mu[3]) * rs[3] * lnh_g[i]         + lnh_b[i];
        float hz = (hrow[H + i]     - mu[4]) * rs[4] * lnh_g[H + i]     + lnh_b[H + i];
        float hn = (hrow[2 * H + i] - mu[5]) * rs[5] * lnh_g[2 * H + i] + lnh_b[2 * H + i];
        float r  = 1.f / (1.f + expf(-(xr + hr)));
        float z  = 1.f / (1.f + expf(-(xz + hz)));
        float nv = tanhf(xn + r * hn);
        float hp = h[(long)b * H + i];
        float hnew = (1.f - z) * nv + z * hp;
        h[(long)b * H + i] = hnew;
        __bf16 bb = (__bf16)hnew;
        hA[(long)b * H + i] = bb;
        if (hA2) hA2[(long)b * H + i] = bb;
    }
}

// ---------------- batched logits GEMM: [S*B x 768] x [768 x V] + bout ----------------
// Hall: bf16 [S*8][768] (row = s*8+b, all rows valid).  WTo: bf16 [V][768].
// out: f32 [B][S][V].
__global__ __launch_bounds__(256) void k_logits(const __bf16* __restrict__ Hall,
                                                const __bf16* __restrict__ WTo,
                                                const float* __restrict__ bout,
                                                float* __restrict__ out) {
    const int K = 768;
    const long Ntiles = V_SZ / 16;  // 2000
    int wave = threadIdx.x >> 5, lane = threadIdx.x & 31;
    long tile = (long)blockIdx.x * 8 + wave;
    long mt = tile / Ntiles;
    long nt = tile - mt * Ntiles;
    if (mt >= (S_SZ * B_SZ) / 16) return;   // 128 M-tiles
    int n0 = (int)(nt << 4);
    int m  = lane & 15;
    int hi = lane >> 4;
    v8f acc = {};
    acc = gemm_k(Hall + (long)(mt * 16 + m) * K + hi * 8,
                 WTo + (long)(n0 + m) * K + hi * 8, K, acc);
    int col = n0 + m;
    float bv = bout[col];
    int rbase = hi * 8;
#pragma unroll
    for (int r = 0; r < 8; ++r) {
        long grow = mt * 16 + rbase + r;   // = s*8 + b
        long s = grow >> 3, b = grow & 7;
        out[((long)b * S_SZ + s) * V_SZ + col] = acc[r] + bv;
    }
}

// ---------------- host orchestration ----------------
extern "C" void kernel_launch(void* const* d_in, const int* in_sizes, int n_in,
                              void* d_out, int out_size, void* d_ws, size_t ws_size,
                              hipStream_t stream) {
    const int*   ids       = (const int*)  d_in[0];
    const float* emb       = (const float*)d_in[1];
    const float* ln_emb_g  = (const float*)d_in[2];
    const float* ln_emb_b  = (const float*)d_in[3];
    const float* low_Wi    = (const float*)d_in[4];
    const float* low_bi    = (const float*)d_in[5];
    const float* low_Wh    = (const float*)d_in[6];
    const float* low_bh    = (const float*)d_in[7];
    const float* low_lni_g = (const float*)d_in[8];
    const float* low_lni_b = (const float*)d_in[9];
    const float* low_lnh_g = (const float*)d_in[10];
    const float* low_lnh_b = (const float*)d_in[11];
    const float* high_Wi   = (const float*)d_in[12];
    const float* high_bi   = (const float*)d_in[13];
    const float* high_Wh   = (const float*)d_in[14];
    const float* high_bh   = (const float*)d_in[15];
    const float* high_lni_g= (const float*)d_in[16];
    const float* high_lni_b= (const float*)d_in[17];
    const float* high_lnh_g= (const float*)d_in[18];
    const float* high_lnh_b= (const float*)d_in[19];
    const float* Wout      = (const float*)d_in[20];
    const float* bout      = (const float*)d_in[21];
    float* out = (float*)d_out;

    char* ws = (char*)d_ws;
    size_t off = 0;
    auto take = [&](size_t bytes) -> char* {
        char* p = ws + off;
        off = (off + bytes + 255) & ~(size_t)255;
        return p;
    };

    // ---- zero-initialized region (contiguous; all sizes multiples of 256B) ----
    float*  h_l  = (float*) take((size_t)B_SZ * HL_SZ * 4);        //    32768
    float*  h_h  = (float*) take((size_t)B_SZ * HH_SZ * 4);        //    24576
    __bf16* A_hl = (__bf16*)take((size_t)16 * HL_SZ * 2);          //    32768 (16 rows)
    __bf16* A_hh = (__bf16*)take((size_t)16 * HH_SZ * 2);          //    24576 (16 rows)
    __bf16* Xb   = (__bf16*)take((size_t)S_SZ * 16 * E_SZ * 2);    //  6291456 (16 rows/step)
    const size_t zeroBytes = 32768 + 24576 + 32768 + 24576 + (size_t)S_SZ * 16 * E_SZ * 2;
    // ---- rest of workspace ----
    float*  xi_raw = (float*)take((size_t)16 * 3 * HL_SZ * 4);
    float*  hi_raw = (float*)take((size_t)16 * 3 * HL_SZ * 4);
    __bf16* Hall = (__bf16*)take((size_t)S_SZ * B_SZ * HH_SZ * 2);
    __bf16* WTli = (__bf16*)take((size_t)3 * HL_SZ * (E_SZ + HH_SZ) * 2);
    __bf16* WTlh = (__bf16*)take((size_t)3 * HL_SZ * HL_SZ * 2);
    __bf16* WThi = (__bf16*)take((size_t)3 * HH_SZ * HL_SZ * 2);
    __bf16* WThh = (__bf16*)take((size_t)3 * HH_SZ * HH_SZ * 2);
    __bf16* WTo  = (__bf16*)take((size_t)V_SZ * HH_SZ * 2);

    // zero states + padded-row activation buffers
    long zeroWords = (long)(zeroBytes / 4);
    k_zero<<<(int)((zeroWords + 255) / 256), 256, 0, stream>>>((unsigned int*)ws, zeroWords);

    // one-time weight convert+transpose to bf16 (held resident in 192MB L2 thereafter)
    k_w2bf16t<<<(3 * 1536 * 1024) / 256, 256, 0, stream>>>(low_Wi, WTli, 1536, HL_SZ, 3);
    k_w2bf16t<<<(3 * 1024 * 1024) / 256, 256, 0, stream>>>(low_Wh, WTlh, HL_SZ, HL_SZ, 3);
    k_w2bf16t<<<(3 * 1024 * 768) / 256, 256, 0, stream>>>(high_Wi, WThi, HL_SZ, HH_SZ, 3);
    k_w2bf16t<<<(3 * 768 * 768) / 256, 256, 0, stream>>>(high_Wh, WThh, HH_SZ, HH_SZ, 3);
    k_w2bf16t<<<(768 * 32000) / 256, 256, 0, stream>>>(Wout, WTo, HH_SZ, V_SZ, 1);

    // all timestep embeddings + LN up front
    k_embed<<<S_SZ * 8, 256, 0, stream>>>(ids, emb, ln_emb_g, ln_emb_b, Xb);

    // sequential recurrence (kernel boundaries = grid sync)
    for (int s = 0; s < S_SZ; ++s) {
        const __bf16* Xs = Xb + (size_t)s * 16 * E_SZ;
        for (int n = 0; n < 3; ++n) {
            for (int t = 0; t < 5; ++t) {
                // low GRU: xi = [x | h_h] @ low_Wi  and  hi = h_l @ low_Wh
                // (192+192 tiles) / 8 waves = 48 blocks
                k_gru_gemms<<<48, 256, 0, stream>>>(
                    Xs, E_SZ, A_hh, HH_SZ, WTli, low_bi, xi_raw,
                    A_hl, HL_SZ, WTlh, low_bh, hi_raw, 3 * HL_SZ);
                k_gru_update<<<8, 256, 0, stream>>>(xi_raw, hi_raw,
                                                    low_lni_g, low_lni_b,
                                                    low_lnh_g, low_lnh_b,
                                                    h_l, A_hl, (__bf16*)nullptr, HL_SZ);
            }
            // high GRU: xi = h_l @ high_Wi  and  hi = h_h @ high_Wh
            // (144+144 tiles) / 8 waves = 36 blocks
            k_gru_gemms<<<36, 256, 0, stream>>>(
                A_hl, HL_SZ, A_hl, 0, WThi, high_bi, xi_raw,
                A_hh, HH_SZ, WThh, high_bh, hi_raw, 3 * HH_SZ);
            k_gru_update<<<8, 256, 0, stream>>>(xi_raw, hi_raw,
                                                high_lni_g, high_lni_b,
                                                high_lnh_g, high_lnh_b,
                                                h_h, A_hh,
                                                Hall + (size_t)s * B_SZ * HH_SZ, HH_SZ);
        }
    }

    // batched logits: 128 M-tiles x 2000 N-tiles = 256000 tiles / 8 waves = 32000 blocks
    k_logits<<<32000, 256, 0, stream>>>(Hall, WTo, bout, out);
}
